// SelfAttention_74079595921800
// MI455X (gfx1250) — compile-verified, matrix operand order checked by reading, not run on hardware
//
#include <hip/hip_runtime.h>
#include <hip/hip_bf16.h>

// ---------------------------------------------------------------------------
// Self-attention forward for MI455X (gfx1250, wave32, WMMA).
//  Kernel 1: QKV projection GEMM (f32 in -> f16 Q/K/V in workspace).
//  Kernel 2: flash-attention over 64-row q tiles, online softmax.
// All WMMA fragments are built from contiguous 128-bit LDS/global loads;
// B-operand tiles (W, V) are stored TRANSPOSED in LDS. The attention K tile
// (pure f16 byte copy) is staged with GLOBAL_LOAD_ASYNC_TO_LDS_B128
// (ASYNCcnt path) — builtin signature: (v4i AS1*, v4i AS3*, imm, imm).
// ---------------------------------------------------------------------------

typedef _Float16 half_t;
typedef __attribute__((ext_vector_type(16))) _Float16 v16h;
typedef __attribute__((ext_vector_type(8)))  _Float16 v8h;
typedef __attribute__((ext_vector_type(8)))  float    v8f;
typedef __attribute__((ext_vector_type(4)))  int      v4i;

constexpr int kB  = 2;
constexpr int kS  = 2048;
constexpr int kH  = 768;
constexpr int kNH = 12;
constexpr int kHD = 64;
constexpr int kBS = kB * kS;          // 4096 rows of hidden_states

#if defined(__has_builtin)
#if __has_builtin(__builtin_amdgcn_global_load_async_to_lds_b128)
#define USE_ASYNC_LDS 1
#endif
#endif

#ifdef USE_ASYNC_LDS
typedef __attribute__((address_space(1))) v4i gas_v4i;   // global (AS1)
typedef __attribute__((address_space(3))) v4i las_v4i;   // LDS    (AS3)
// Per-lane async copy of 16 bytes global -> LDS (tracked by ASYNCcnt).
__device__ __forceinline__ void async_copy16(const half_t* gsrc, half_t* ldst) {
  __builtin_amdgcn_global_load_async_to_lds_b128((gas_v4i*)gsrc,
                                                 (las_v4i*)ldst, 0, 0);
}
__device__ __forceinline__ void wait_asynccnt0() {
#if __has_builtin(__builtin_amdgcn_s_wait_asynccnt)
  __builtin_amdgcn_s_wait_asynccnt(0);
#else
  asm volatile("s_wait_asynccnt 0" ::: "memory");
#endif
}
#endif

union frag16 { v16h v; v8h h[2]; };

// Build a 16-half fragment from two aligned 8-half (128-bit) loads.
__device__ __forceinline__ v16h ld_frag(const half_t* lo, const half_t* hi) {
  frag16 f;
  f.h[0] = *(const v8h*)lo;
  f.h[1] = *(const v8h*)hi;
  return f.v;
}

// 16-lane-group reductions (C/D layout: lanes 0..15 / 16..31 are independent
// row groups; xor masks 1,2,4,8 never cross the lane-16 boundary). wave32.
__device__ __forceinline__ float redmax16(float v) {
  v = fmaxf(v, __shfl_xor(v, 1, 32));
  v = fmaxf(v, __shfl_xor(v, 2, 32));
  v = fmaxf(v, __shfl_xor(v, 4, 32));
  v = fmaxf(v, __shfl_xor(v, 8, 32));
  return v;
}
__device__ __forceinline__ float redsum16(float v) {
  v += __shfl_xor(v, 1, 32);
  v += __shfl_xor(v, 2, 32);
  v += __shfl_xor(v, 4, 32);
  v += __shfl_xor(v, 8, 32);
  return v;
}

__device__ __forceinline__ v8f wmma16(v16h a, v16h b, v8f c) {
  // D = A(16x32 f16) * B(32x16 f16) + C(16x16 f32)
  return __builtin_amdgcn_wmma_f32_16x16x32_f16(false, a, false, b,
                                                (short)0, c, false, false);
}

// ---------------------------------------------------------------------------
// Kernel 1: O = f16(X @ W + b).  X:[4096,768] f32, W:[768,768] f32 (row = in).
// grid = (64 M-tiles, 12 N-tiles, 3 {Q,K,V}), block = 128 (4 waves).
// ---------------------------------------------------------------------------
__global__ __launch_bounds__(128)
void qkv_gemm_kernel(const float* __restrict__ X,
                     const float* __restrict__ Wq, const float* __restrict__ bq,
                     const float* __restrict__ Wk, const float* __restrict__ bk,
                     const float* __restrict__ Wv, const float* __restrict__ bv,
                     half_t* __restrict__ Q, half_t* __restrict__ K,
                     half_t* __restrict__ V)
{
  const float* Wm;  const float* bias;  half_t* O;
  if (blockIdx.z == 0)      { Wm = Wq; bias = bq; O = Q; }
  else if (blockIdx.z == 1) { Wm = Wk; bias = bk; O = K; }
  else                      { Wm = Wv; bias = bv; O = V; }

  const int M0   = blockIdx.x * 64;
  const int N0   = blockIdx.y * 64;
  const int tid  = threadIdx.x;
  const int wave = tid >> 5;
  const int lane = tid & 31;
  const int l16  = lane & 15;
  const int hi   = lane >> 4;       // lane-group (0 | 1)
  const int koff = hi * 8;          // contraction offset for upper group

  // Row strides 40 halves = 80 B (multiple of 16 B) -> aligned b128 access.
  __shared__ alignas(16) half_t Xs [64][40];  // X tile, row-major  [m][k]
  __shared__ alignas(16) half_t Wst[64][40];  // W tile, TRANSPOSED [n][k]

  const v8f zero = {0.f,0.f,0.f,0.f,0.f,0.f,0.f,0.f};
  v8f acc[4];
  #pragma unroll
  for (int j = 0; j < 4; ++j) acc[j] = zero;

  for (int k0 = 0; k0 < kH; k0 += 32) {
    { // stage X tile (64 rows x 32 k) as f16; float4 global reads
      const int r = tid >> 1, c = (tid & 1) * 16;
      const float4* src = (const float4*)(X + (size_t)(M0 + r) * kH + k0 + c);
      #pragma unroll
      for (int i = 0; i < 4; ++i) {
        const float4 f = src[i];
        Xs[r][c + 4*i + 0] = (half_t)f.x;
        Xs[r][c + 4*i + 1] = (half_t)f.y;
        Xs[r][c + 4*i + 2] = (half_t)f.z;
        Xs[r][c + 4*i + 3] = (half_t)f.w;
      }
    }
    { // stage W tile (32 k x 64 n) transposed -> Wst[n][k]
      const int r = tid >> 2, c = (tid & 3) * 16;   // r = k row, c = n base
      const float4* src = (const float4*)(Wm + (size_t)(k0 + r) * kH + N0 + c);
      #pragma unroll
      for (int i = 0; i < 4; ++i) {
        const float4 f = src[i];
        Wst[c + 4*i + 0][r] = (half_t)f.x;
        Wst[c + 4*i + 1][r] = (half_t)f.y;
        Wst[c + 4*i + 2][r] = (half_t)f.z;
        Wst[c + 4*i + 3][r] = (half_t)f.w;
      }
    }
    __syncthreads();

    // A fragment (rows 16*wave..+15): contiguous 8-half runs -> 2x b128
    const int m = 16 * wave + l16;
    const v16h a = ld_frag(&Xs[m][koff], &Xs[m][16 + koff]);

    // B fragments for 4 N sub-tiles: transposed tile -> 2x b128 each
    #pragma unroll
    for (int j = 0; j < 4; ++j) {
      const int n = 16 * j + l16;
      const v16h bfr = ld_frag(&Wst[n][koff], &Wst[n][16 + koff]);
      acc[j] = wmma16(a, bfr, acc[j]);
    }
    __syncthreads();
  }

  // Epilogue: C/D layout -> global f16 with bias.
  #pragma unroll
  for (int j = 0; j < 4; ++j) {
    const int n = N0 + 16 * j + l16;
    const float bv_ = bias[n];
    #pragma unroll
    for (int r = 0; r < 8; ++r) {
      const int m2 = M0 + 16 * wave + r + 8 * hi;
      O[(size_t)m2 * kH + n] = (half_t)(acc[j][r] + bv_);
    }
  }
}

// ---------------------------------------------------------------------------
// Kernel 2: flash attention.  grid = (S/64 q-tiles, NH, B), block = 128.
// Each wave owns 16 q rows; k loop in steps of 32.
// ---------------------------------------------------------------------------
__global__ __launch_bounds__(128)
void attention_kernel(const half_t* __restrict__ Q,
                      const half_t* __restrict__ K,
                      const half_t* __restrict__ V,
                      float* __restrict__ out)
{
  const int qt = blockIdx.x;        // q tile (64 rows)
  const int h  = blockIdx.y;        // head
  const int b  = blockIdx.z;        // batch

  const int tid  = threadIdx.x;
  const int wave = tid >> 5;
  const int lane = tid & 31;
  const int l16  = lane & 15;
  const int hi   = lane >> 4;
  const int koff = hi * 8;

  const size_t headOff = (size_t)b * kS * kH + (size_t)h * kHD;
  const half_t* Qb = Q + headOff;   // row stride kH
  const half_t* Kb = K + headOff;
  const half_t* Vb = V + headOff;

  __shared__ alignas(16) half_t Ks [32][72];  // K tile row-major [k][d], 144 B rows
  __shared__ alignas(16) half_t Vst[64][40];  // V tile TRANSPOSED [d][k],  80 B rows
  __shared__ alignas(16) half_t Ps [4][16][40]; // per-wave P staging (16 q x 32 k)

  // Q A-fragments for this wave, d chunks [0,32) and [32,64): 4x global b128.
  const int q0 = qt * 64 + wave * 16;
  const half_t* qrow = Qb + (size_t)(q0 + l16) * kH;
  const v16h aq0 = ld_frag(qrow +      koff, qrow + 16 + koff);
  const v16h aq1 = ld_frag(qrow + 32 + koff, qrow + 48 + koff);

  const v8f zero = {0.f,0.f,0.f,0.f,0.f,0.f,0.f,0.f};
  v8f cacc[4];
  #pragma unroll
  for (int j = 0; j < 4; ++j) cacc[j] = zero;
  float mrow[8], lsum[8];
  #pragma unroll
  for (int r = 0; r < 8; ++r) { mrow[r] = -1.0e30f; lsum[r] = 0.0f; }

  const float scale = 0.125f;       // 1/sqrt(64)

  for (int k0 = 0; k0 < kS; k0 += 32) {
    { // stage K (row-major, async DMA to LDS) and V (transposed, via VGPRs)
      const int r = tid >> 2, c = (tid & 3) * 16;   // r = k row, c = d base
      const half_t* ksrc = Kb + (size_t)(k0 + r) * kH + c;
      const half_t* vsrc = Vb + (size_t)(k0 + r) * kH + c;
#ifdef USE_ASYNC_LDS
      async_copy16(ksrc,     &Ks[r][c]);      // global_load_async_to_lds_b128
      async_copy16(ksrc + 8, &Ks[r][c + 8]);
#else
      *(v8h*)&Ks[r][c]     = *(const v8h*)(ksrc);
      *(v8h*)&Ks[r][c + 8] = *(const v8h*)(ksrc + 8);
#endif
      const v8h vv0 = *(const v8h*)(vsrc);
      const v8h vv1 = *(const v8h*)(vsrc + 8);
      #pragma unroll
      for (int i = 0; i < 8; ++i) {
        Vst[c + i][r]     = vv0[i];
        Vst[c + 8 + i][r] = vv1[i];
      }
      if (k0 + 32 < kS) {           // hint next tiles toward cache
        __builtin_prefetch(ksrc + 32 * kH, 0, 1);
        __builtin_prefetch(vsrc + 32 * kH, 0, 1);
      }
    }
#ifdef USE_ASYNC_LDS
    wait_asynccnt0();               // s_wait_asynccnt 0: K tile landed in LDS
#endif
    __syncthreads();

    // scores: S = Q * K^T  (B column = k index, contraction = d)
    v8f s[2];
    #pragma unroll
    for (int j = 0; j < 2; ++j) {
      const int kcol = 16 * j + l16;
      v8f acc_s = zero;
      const v16h bk0 = ld_frag(&Ks[kcol][koff],      &Ks[kcol][16 + koff]);
      const v16h bk1 = ld_frag(&Ks[kcol][32 + koff], &Ks[kcol][48 + koff]);
      acc_s = wmma16(aq0, bk0, acc_s);
      acc_s = wmma16(aq1, bk1, acc_s);
      s[j] = acc_s;
    }

    // Online softmax per q-row; write P (f16) into per-wave LDS staging.
    #pragma unroll
    for (int r = 0; r < 8; ++r) {
      const float s0 = s[0][r] * scale;
      const float s1 = s[1][r] * scale;
      const float tmax = redmax16(fmaxf(s0, s1));
      const float mnew = fmaxf(mrow[r], tmax);
      const float corr = __expf(mrow[r] - mnew);
      const float p0 = __expf(s0 - mnew);
      const float p1 = __expf(s1 - mnew);
      lsum[r] = lsum[r] * corr + redsum16(p0 + p1);
      mrow[r] = mnew;
      #pragma unroll
      for (int j = 0; j < 4; ++j) cacc[j][r] *= corr;
      const int m = r + 8 * hi;
      Ps[wave][m][l16]      = (half_t)p0;
      Ps[wave][m][16 + l16] = (half_t)p1;
    }
    __syncthreads();   // P visible (and all Ks reads retired)

    // ctx += P(16x32) * V(32x64): A from Ps, B from transposed Vst.
    const v16h ap = ld_frag(&Ps[wave][l16][koff], &Ps[wave][l16][16 + koff]);
    #pragma unroll
    for (int j = 0; j < 4; ++j) {
      const int dcol = 16 * j + l16;
      const v16h bvf = ld_frag(&Vst[dcol][koff], &Vst[dcol][16 + koff]);
      cacc[j] = wmma16(ap, bvf, cacc[j]);
    }
    __syncthreads();   // all Vst/Ps reads done before next tile load
  }

  // Normalize and store ctx (f32). Output layout [B,S,NH*HD]: head h occupies
  // columns h*64..h*64+63.
  #pragma unroll
  for (int r = 0; r < 8; ++r) {
    const float inv = 1.0f / lsum[r];
    const int q = q0 + r + 8 * hi;
    float* orow = out + (size_t)(b * kS + q) * kH + (size_t)h * kHD;
    #pragma unroll
    for (int j = 0; j < 4; ++j)
      orow[16 * j + l16] = cacc[j][r] * inv;
  }
}

// ---------------------------------------------------------------------------
extern "C" void kernel_launch(void* const* d_in, const int* in_sizes, int n_in,
                              void* d_out, int out_size, void* d_ws, size_t ws_size,
                              hipStream_t stream) {
  const float* X  = (const float*)d_in[0];
  const float* Wq = (const float*)d_in[1];
  const float* bq = (const float*)d_in[2];
  const float* Wk = (const float*)d_in[3];
  const float* bk = (const float*)d_in[4];
  const float* Wv = (const float*)d_in[5];
  const float* bv = (const float*)d_in[6];
  float* out = (float*)d_out;

  // Workspace: f16 Q, K, V, each [4096, 768] (6.29 MB each, 18.9 MB total).
  half_t* Qh = (half_t*)d_ws;
  half_t* Kh = Qh + (size_t)kBS * kH;
  half_t* Vh = Kh + (size_t)kBS * kH;

  dim3 g1(kBS / 64, kH / 64, 3);       // 64 x 12 x 3 blocks
  qkv_gemm_kernel<<<g1, 128, 0, stream>>>(X, Wq, bq, Wk, bk, Wv, bv, Qh, Kh, Vh);

  dim3 g2(kS / 64, kNH, kB);           // 32 x 12 x 2 blocks
  attention_kernel<<<g2, 128, 0, stream>>>(Qh, Kh, Vh, out);
}
